// GATEncoder_19756849561858
// MI455X (gfx1250) — matrix operand order repaired and checked by reading, not compiled, not run on hardware
//
#include <hip/hip_runtime.h>

#define DHC 256   // D*H
#define HC  4     // heads
#define DC  64    // per-head dim
#define NEG_SLOPE 0.1f

typedef __attribute__((ext_vector_type(16))) __bf16 v16bf;
typedef __attribute__((ext_vector_type(8)))  float  v8f;
typedef __attribute__((ext_vector_type(4)))  unsigned int v4u;
typedef __attribute__((ext_vector_type(8)))  int v8i;
typedef __attribute__((ext_vector_type(4)))  int v4i;

__device__ __forceinline__ unsigned short f2bf(float f) {
  unsigned u = __float_as_uint(f);
  unsigned r = 0x7FFFu + ((u >> 16) & 1u);     // round-to-nearest-even
  return (unsigned short)((u + r) >> 16);
}
// monotone float<->uint mapping so unsigned atomicMax == float max
__device__ __forceinline__ unsigned fenc(float f) {
  unsigned u = __float_as_uint(f);
  return (u & 0x80000000u) ? ~u : (u | 0x80000000u);
}
__device__ __forceinline__ float fdec(unsigned u) {
  return (u & 0x80000000u) ? __uint_as_float(u & 0x7FFFFFFFu) : __uint_as_float(~u);
}

// ---------------- prep: W[k][j] (f32) -> Wt[j][k] (bf16, column-major for B frags)
__global__ void k_prep_w(const float* __restrict__ W, unsigned short* __restrict__ Wt) {
  int tid = blockIdx.x * blockDim.x + threadIdx.x;   // 65536
  int j = tid >> 8, k = tid & 255;
  Wt[tid] = f2bf(W[k * DHC + j]);
}

// ---------------- embedding lookup: x = W_pre_t[n_id] + b_pre  (stored bf16)
__global__ void k_embed(const int* __restrict__ nid, const float* __restrict__ Wpre,
                        const float* __restrict__ bpre, unsigned short* __restrict__ xbf,
                        int N) {
  int tid = blockIdx.x * blockDim.x + threadIdx.x;
  if (tid >= N * DHC) return;
  int n = tid >> 8, c = tid & 255;
  long long row = nid[n];
  xbf[tid] = f2bf(Wpre[row * (long long)DHC + c] + bpre[c]);
}

// ---------------- dual GEMM: h_src = x@W_src, h_dst = x@W_dst  (bf16 WMMA, f32 acc)
// A tile (64 rows x 256 K bf16) staged into LDS by the Tensor Data Mover with
// hardware row padding (512B row + 16B pad -> 528B stride, bank-conflict free).
__global__ __launch_bounds__(256) void k_gemm_dual(
    const unsigned short* __restrict__ xbf,
    const unsigned short* __restrict__ wsrc_t,
    const unsigned short* __restrict__ wdst_t,
    float* __restrict__ hsrc, float* __restrict__ hdst, int N) {
  __shared__ unsigned short Atile[64 * 264];   // 64 rows, padded stride 264 ushorts
  const int tid = threadIdx.x;
  const int blk = blockIdx.x;

#if __has_builtin(__builtin_amdgcn_tensor_load_to_lds) && __has_builtin(__builtin_amdgcn_s_wait_tensorcnt)
  if (tid < 32) {                              // wave 0 issues one TDM DMA per block
    unsigned long long ga = (unsigned long long)(uintptr_t)xbf
                          + (unsigned long long)blk * (64ull * DHC * 2ull);
    unsigned ldsb = (unsigned)(uintptr_t)(void*)&Atile[0];   // low 32b = LDS byte offset
    unsigned rows_rem = (unsigned)(N - blk * 64);            // OOB rows read as zero
    v4u g0;
    g0.x = 1u;                                               // count=1 (valid descriptor)
    g0.y = ldsb;                                             // lds_addr
    g0.z = (unsigned)(ga & 0xFFFFFFFFull);                   // global_addr[31:0]
    g0.w = (unsigned)((ga >> 32) & 0x01FFFFFFull) | (2u << 30); // addr[56:32] | type=2
    v8i g1;
    // data_size=1 (2B) | pad_enable | pad_interval=6 (128 dw = 512B) | pad_amount=3 (4 dw = 16B)
    g1[0] = (int)((1u << 16) | (1u << 20) | (6u << 22) | (3u << 25));
    g1[1] = (int)(256u << 16);                               // tensor_dim0[15:0]=256 @ bits63:48
    g1[2] = (int)((rows_rem & 0xFFFFu) << 16);               // dim0 hi=0 | tensor_dim1[15:0]
    g1[3] = (int)(((rows_rem >> 16) & 0xFFFFu) | (256u << 16)); // dim1 hi | tile_dim0=256
    g1[4] = 64;                                              // tile_dim1=64 | tile_dim2=0
    g1[5] = 256;                                             // tensor_dim0_stride[31:0]
    g1[6] = 0;                                               // stride hi | dim1_stride lo
    g1[7] = 0;
    v4i g2 = {0, 0, 0, 0};
    v4i g3 = {0, 0, 0, 0};
    v8i g4 = {0, 0, 0, 0, 0, 0, 0, 0};
    __builtin_amdgcn_tensor_load_to_lds(g0, g1, g2, g3, g4, 0);
    __builtin_amdgcn_s_wait_tensorcnt(0);
  }
#else
  for (int i = 0; i < 8; ++i) {
    int chunk = tid + i * 256;
    int row = chunk >> 5;
    int kc  = (chunk & 31) << 3;
    int grow = blk * 64 + row; if (grow > N - 1) grow = N - 1;
    *(float4*)&Atile[row * 264 + kc] = *(const float4*)&xbf[(size_t)grow * DHC + kc];
  }
#endif
  __syncthreads();

  const int wave = tid >> 5;
  const int lane = tid & 31;
  const int half = lane >> 4;   // 0: lanes 0-15, 1: lanes 16-31
  const int lrow = lane & 15;

  for (int t = wave; t < 64; t += 8) {          // 4x16 grid of 16x16 output tiles
    const int msub  = t >> 4;
    const int ntile = t & 15;
    v8f cs = {0.f,0.f,0.f,0.f,0.f,0.f,0.f,0.f};
    v8f cd = {0.f,0.f,0.f,0.f,0.f,0.f,0.f,0.f};
    const int arow = msub * 16 + lrow;
    const int bcol = ntile * 16 + lrow;
#pragma unroll
    for (int k0 = 0; k0 < DHC; k0 += 32) {
      union { v16bf v; float4 f[2]; } a, bs, bd;
      // A frag: lanes<16 hold K {0..7,16..23}, lanes>=16 hold K {8..15,24..31}
      const unsigned short* ap = &Atile[arow * 264 + k0 + half * 8];
      a.f[0] = *(const float4*)ap;
      a.f[1] = *(const float4*)(ap + 16);
      // B frag: lanes<16 hold K {0..7,16..23}, lanes>=16 hold K {8..15,24..31} (col-major Wt)
      const unsigned short* bsp = &wsrc_t[(size_t)bcol * DHC + k0 + half * 8];
      bs.f[0] = *(const float4*)bsp;
      bs.f[1] = *(const float4*)(bsp + 16);
      const unsigned short* bdp = &wdst_t[(size_t)bcol * DHC + k0 + half * 8];
      bd.f[0] = *(const float4*)bdp;
      bd.f[1] = *(const float4*)(bdp + 16);
      cs = __builtin_amdgcn_wmma_f32_16x16x32_bf16(false, a.v, false, bs.v, (short)0, cs, false, false);
      cd = __builtin_amdgcn_wmma_f32_16x16x32_bf16(false, a.v, false, bd.v, (short)0, cd, false, false);
    }
    // C layout: lanes<16 -> M=r, lanes>=16 -> M=8+r; N = lane%16
#pragma unroll
    for (int r = 0; r < 8; ++r) {
      int rowg = blk * 64 + msub * 16 + half * 8 + r;
      if (rowg < N) {
        size_t idx = (size_t)rowg * DHC + ntile * 16 + lrow;
        hsrc[idx] = cs[r];
        hdst[idx] = cd[r];
      }
    }
  }
}

// ---------------- per-node attention scores a_src/a_dst [N,H]
__global__ void k_att(const float* __restrict__ hsrc, const float* __restrict__ hdst,
                      const float* __restrict__ att_s, const float* __restrict__ att_d,
                      float* __restrict__ a_src, float* __restrict__ a_dst, int N) {
  int tid = blockIdx.x * blockDim.x + threadIdx.x;
  if (tid >= N * HC) return;
  int n = tid >> 2, h = tid & 3;
  const float* ps = hsrc + (size_t)n * DHC + h * DC;
  const float* pd = hdst + (size_t)n * DHC + h * DC;
  const float* vs = att_s + h * DC;
  const float* vd = att_d + h * DC;
  float ss = 0.f, sd = 0.f;
#pragma unroll 8
  for (int d = 0; d < DC; ++d) { ss += ps[d] * vs[d]; sd += pd[d] * vd[d]; }
  a_src[tid] = ss; a_dst[tid] = sd;
}

__global__ void k_zero(float* __restrict__ p, long long n) {
  long long tid = (long long)blockIdx.x * blockDim.x + threadIdx.x;
  if (tid < n) p[tid] = 0.f;
}

// ---------------- pass 1: e = leaky_relu(a_src[src]+a_dst[dst]); segment max -> m_enc
__global__ void k_edge_att(const int* __restrict__ esrc, const int* __restrict__ edst,
                           int E, int N, const float* __restrict__ as,
                           const float* __restrict__ ad, float* __restrict__ ebuf,
                           unsigned* __restrict__ menc) {
  int tid = blockIdx.x * blockDim.x + threadIdx.x;
  if (tid >= E + N) return;
  int s = (tid < E) ? esrc[tid] : (tid - E);
  int d = (tid < E) ? edst[tid] : (tid - E);
  float4 v;
  float* vp = &v.x;
#pragma unroll
  for (int h = 0; h < HC; ++h) {
    float e = as[s * HC + h] + ad[d * HC + h];
    e = (e > 0.f) ? e : NEG_SLOPE * e;
    vp[h] = e;
    atomicMax(&menc[d * HC + h], fenc(e));
  }
  *(float4*)(ebuf + (size_t)tid * HC) = v;
}

// ---------------- pass 2: p = exp(e - m[dst]); segment sum -> s
__global__ void k_edge_exp(const int* __restrict__ edst, int E, int N,
                           float* __restrict__ ebuf, const unsigned* __restrict__ menc,
                           float* __restrict__ ssum) {
  int tid = blockIdx.x * blockDim.x + threadIdx.x;
  if (tid >= E + N) return;
  int d = (tid < E) ? edst[tid] : (tid - E);
  float4 e = *(float4*)(ebuf + (size_t)tid * HC);
  uint4 m  = *(const uint4*)(menc + d * HC);
  float4 p;
  p.x = __expf(e.x - fdec(m.x));
  p.y = __expf(e.y - fdec(m.y));
  p.z = __expf(e.z - fdec(m.z));
  p.w = __expf(e.w - fdec(m.w));
  atomicAdd(&ssum[d * HC + 0], p.x);
  atomicAdd(&ssum[d * HC + 1], p.y);
  atomicAdd(&ssum[d * HC + 2], p.z);
  atomicAdd(&ssum[d * HC + 3], p.w);
  *(float4*)(ebuf + (size_t)tid * HC) = p;
}

// ---------------- pass 2b: attn = p / s[dst] * w_edge
__global__ void k_edge_norm(const int* __restrict__ edst, const float* __restrict__ w,
                            int E, int N, float* __restrict__ ebuf,
                            const float* __restrict__ ssum) {
  int tid = blockIdx.x * blockDim.x + threadIdx.x;
  if (tid >= E + N) return;
  int d = (tid < E) ? edst[tid] : (tid - E);
  float we = (tid < E) ? w[tid] : 1.0f;
  float4 p = *(float4*)(ebuf + (size_t)tid * HC);
  float4 s = *(const float4*)(ssum + d * HC);
  p.x = p.x / s.x * we; p.y = p.y / s.y * we;
  p.z = p.z / s.z * we; p.w = p.w / s.w * we;
  *(float4*)(ebuf + (size_t)tid * HC) = p;
}

// ---------------- pass 3: out[dst] += h_src[src] * attn  (64 threads per edge)
__global__ void k_edge_scatter(const int* __restrict__ esrc, const int* __restrict__ edst,
                               int E, int N, const float* __restrict__ hsrc,
                               const float* __restrict__ ebuf, float* __restrict__ out) {
  long long gtid = (long long)blockIdx.x * blockDim.x + threadIdx.x;
  long long total = (long long)(E + N) * DC;
  if (gtid >= total) return;
  int e = (int)(gtid >> 6);
  int d = (int)(gtid & 63);
  int ss = (e < E) ? esrc[e] : (e - E);
  int dd = (e < E) ? edst[e] : (e - E);
  float4 a = *(const float4*)(ebuf + (size_t)e * HC);
  const float* hp = hsrc + (size_t)ss * DHC + d;
  float* op = out + (size_t)dd * DHC + d;
  atomicAdd(op + 0 * DC, hp[0 * DC] * a.x);
  atomicAdd(op + 1 * DC, hp[1 * DC] * a.y);
  atomicAdd(op + 2 * DC, hp[2 * DC] * a.z);
  atomicAdd(op + 3 * DC, hp[3 * DC] * a.w);
}

// ---------------- finish hop 1: x1 = out + bias (f32 + bf16 copies)
__global__ void k_finish1(const float* __restrict__ out, const float* __restrict__ bias,
                          float* __restrict__ x1, unsigned short* __restrict__ xbf, int N) {
  int tid = blockIdx.x * blockDim.x + threadIdx.x;
  if (tid >= N * DHC) return;
  float v = out[tid] + bias[tid & 255];
  x1[tid] = v;
  xbf[tid] = f2bf(v);
}

// ---------------- finish hop 2: result = x1 + out2 + bias
__global__ void k_finish2(const float* __restrict__ out, const float* __restrict__ bias,
                          const float* __restrict__ x1, float* __restrict__ res, int N) {
  int tid = blockIdx.x * blockDim.x + threadIdx.x;
  if (tid >= N * DHC) return;
  res[tid] = x1[tid] + out[tid] + bias[tid & 255];
}

static inline size_t al256(size_t x) { return (x + 255) & ~(size_t)255; }

extern "C" void kernel_launch(void* const* d_in, const int* in_sizes, int n_in,
                              void* d_out, int out_size, void* d_ws, size_t ws_size,
                              hipStream_t stream) {
  const int*   nid   = (const int*)d_in[0];
  const int*   ei1   = (const int*)d_in[1];
  const float* w1    = (const float*)d_in[2];
  const int*   ei2   = (const int*)d_in[3];
  const float* w2    = (const float*)d_in[4];
  const float* Wpre  = (const float*)d_in[5];
  const float* bpre  = (const float*)d_in[6];
  const float* Wsrc  = (const float*)d_in[7];
  const float* Wdst  = (const float*)d_in[8];
  const float* att_s = (const float*)d_in[9];
  const float* att_d = (const float*)d_in[10];
  const float* bias  = (const float*)d_in[11];
  const int N = in_sizes[0];
  const int E = in_sizes[1] / 2;

  char* ws = (char*)d_ws;
  size_t off = 0;
  unsigned short* wsrc_t = (unsigned short*)(ws + off); off += al256((size_t)DHC * DHC * 2);
  unsigned short* wdst_t = (unsigned short*)(ws + off); off += al256((size_t)DHC * DHC * 2);
  unsigned short* xbf    = (unsigned short*)(ws + off); off += al256((size_t)N * DHC * 2);
  float* x1    = (float*)(ws + off); off += al256((size_t)N * DHC * 4);
  float* hsrc  = (float*)(ws + off); off += al256((size_t)N * DHC * 4);
  float* hdst  = (float*)(ws + off); off += al256((size_t)N * DHC * 4);
  float* out   = (float*)(ws + off); off += al256((size_t)N * DHC * 4);   // start of zero region
  float* a_src = (float*)(ws + off); off += al256((size_t)N * HC * 4);
  float* a_dst = (float*)(ws + off); off += al256((size_t)N * HC * 4);
  unsigned* menc = (unsigned*)(ws + off); off += al256((size_t)N * HC * 4);
  float* ssum  = (float*)(ws + off); off += al256((size_t)N * HC * 4);    // end of zero region
  float* ebuf  = (float*)(ws + off); off += al256((size_t)(E + N) * HC * 4);
  (void)n_in; (void)out_size; (void)ws_size;

  const int TB = 256;
  const long long zeroN = (long long)N * DHC + 4LL * N * HC;  // out..ssum contiguous
  const int nodeDHblocks = (N * DHC + TB - 1) / TB;
  const int edgeBlocks = (E + N + TB - 1) / TB;
  const long long scatTotal = (long long)(E + N) * DC;
  const int scatBlocks = (int)((scatTotal + TB - 1) / TB);
  const int gemmBlocks = (N + 63) / 64;

  // one-time prep (cheap; rerun every call for determinism)
  k_prep_w<<<(DHC * DHC) / TB, TB, 0, stream>>>(Wsrc, wsrc_t);
  k_prep_w<<<(DHC * DHC) / TB, TB, 0, stream>>>(Wdst, wdst_t);
  k_embed<<<nodeDHblocks, TB, 0, stream>>>(nid, Wpre, bpre, xbf, N);

  // ---- hop 1
  k_zero<<<(int)((zeroN + TB - 1) / TB), TB, 0, stream>>>(out, zeroN);
  k_gemm_dual<<<gemmBlocks, TB, 0, stream>>>(xbf, wsrc_t, wdst_t, hsrc, hdst, N);
  k_att<<<(N * HC + TB - 1) / TB, TB, 0, stream>>>(hsrc, hdst, att_s, att_d, a_src, a_dst, N);
  k_edge_att<<<edgeBlocks, TB, 0, stream>>>(ei1, ei1 + E, E, N, a_src, a_dst, ebuf, menc);
  k_edge_exp<<<edgeBlocks, TB, 0, stream>>>(ei1 + E, E, N, ebuf, menc, ssum);
  k_edge_norm<<<edgeBlocks, TB, 0, stream>>>(ei1 + E, w1, E, N, ebuf, ssum);
  k_edge_scatter<<<scatBlocks, TB, 0, stream>>>(ei1, ei1 + E, E, N, hsrc, ebuf, out);
  k_finish1<<<nodeDHblocks, TB, 0, stream>>>(out, bias, x1, xbf, N);

  // ---- hop 2
  k_zero<<<(int)((zeroN + TB - 1) / TB), TB, 0, stream>>>(out, zeroN);
  k_gemm_dual<<<gemmBlocks, TB, 0, stream>>>(xbf, wsrc_t, wdst_t, hsrc, hdst, N);
  k_att<<<(N * HC + TB - 1) / TB, TB, 0, stream>>>(hsrc, hdst, att_s, att_d, a_src, a_dst, N);
  k_edge_att<<<edgeBlocks, TB, 0, stream>>>(ei2, ei2 + E, E, N, a_src, a_dst, ebuf, menc);
  k_edge_exp<<<edgeBlocks, TB, 0, stream>>>(ei2 + E, E, N, ebuf, menc, ssum);
  k_edge_norm<<<edgeBlocks, TB, 0, stream>>>(ei2 + E, w2, E, N, ebuf, ssum);
  k_edge_scatter<<<scatBlocks, TB, 0, stream>>>(ei2, ei2 + E, E, N, hsrc, ebuf, out);
  k_finish2<<<nodeDHblocks, TB, 0, stream>>>(out, bias, x1, (float*)d_out, N);
}
